// MotionVectorDeepTracker_4818953306548
// MI455X (gfx1250) — compile-verified, matrix operand order checked by reading, not run on hardware
//
#include <hip/hip_runtime.h>
#include <hip/hip_bf16.h>
#include <math.h>

// ---------------------------------------------------------------------------
// MotionVectorDeepTracker for MI455X (gfx1250).
// NHWC f16 activations; convs are implicit-GEMM on v_wmma_f32_16x16x32_f16
// with DIRECT per-lane b128 fragment loads from global memory (no LDS in the
// hot loop). Weights reordered to k = (kh*KW+kw)*CIN + cin so fragments are
// contiguous. Small numerics (integral image, softmax, LayerNorm) stay fp32.
// Requires ~500 MB of workspace (d_ws).
// ---------------------------------------------------------------------------

typedef _Float16 v16h __attribute__((ext_vector_type(16)));
typedef float    v8f  __attribute__((ext_vector_type(8)));
typedef unsigned int u32x4 __attribute__((ext_vector_type(4)));

#define BATCH 4
#define HH 256
#define WW 256
#define NOBJ 128
#define HID 128
#define HEADS 8
#define FFD 256
#define NTOK (BATCH * NOBJ)   // 512
#define SW 257                 // integral image side (H+1 == W+1)
#define SPLANE (SW * SW)       // 66049

// ============== weight convert + reorder: OIHW f32 -> [co][s*CIN+cin] f16 ==
__global__ void cvt_w_kernel(const float* __restrict__ src,
                             _Float16* __restrict__ dst,
                             int Cout, int CIN, int KHW) {
  const int i = blockIdx.x * 256 + threadIdx.x;
  const int n = Cout * CIN * KHW;
  if (i < n) {
    const int co  = i / (CIN * KHW);
    const int r   = i - co * (CIN * KHW);
    const int s   = r / CIN;
    const int cin = r - s * CIN;
    dst[i] = (_Float16)src[(size_t)co * CIN * KHW + (size_t)cin * KHW + s];
  }
}

// ============== motion vectors: NCHW f32 -> NHWC f16 (CIN = 2) =============
__global__ void mv_to_nhwc_kernel(const float* __restrict__ src,
                                  _Float16* __restrict__ dst) {
  const int p = blockIdx.x * 256 + threadIdx.x;    // pixel in [0, B*H*W)
  const int b = p >> 16;
  const int hw = p & 65535;
  dst[(size_t)p * 2 + 0] = (_Float16)src[((size_t)b * 2 + 0) * 65536 + hw];
  dst[(size_t)p * 2 + 1] = (_Float16)src[((size_t)b * 2 + 1) * 65536 + hw];
}

// ========== fast conv: CIN % 32 == 0, direct b128 fragment loads ===========
// Block = 128 threads (4 waves); tile = 16 Cout x 64 pixels.
// A/B fragment halves are contiguous 16B runs; inner loop = 4 loads + 1 WMMA.
template <int CIN, int KH, int KW, int PAD, bool RELU, typename Tout>
__global__ void conv_fast_kernel(const _Float16* __restrict__ in,
                                 const _Float16* __restrict__ wgt,
                                 const float* __restrict__ bias,
                                 Tout* __restrict__ out,
                                 int CoutTotal, int coBase) {
  static_assert(CIN % 32 == 0, "fast path needs CIN %% 32 == 0");
  constexpr int KDIM = CIN * KH * KW;
  constexpr int NCHUNK = KDIM / 32;

  const int wv   = threadIdx.x >> 5;
  const int lane = threadIdx.x & 31;
  const int m    = lane & 15;
  const int koff = (lane >> 4) << 3;               // 0 or 8

  const int p    = blockIdx.x * 64 + (wv << 4) + m; // pixel (same row per tile)
  const int bImg = p >> 16;
  const int hRow = (p >> 8) & 255;
  const int wCol = p & 255;
  const int coTile = blockIdx.y * 16;

  const _Float16* aptr = wgt + (size_t)(coTile + m) * KDIM;

  v8f acc;
#pragma unroll
  for (int i = 0; i < 8; ++i) acc[i] = 0.0f;

#pragma unroll 2
  for (int kc = 0; kc < NCHUNK; ++kc) {
    const int k0 = kc * 32;
    // ---- A fragment: weights, contiguous ----
    v16h a;
    *(u32x4*)&a         = *(const u32x4*)(aptr + k0 + koff);
    *(((u32x4*)&a) + 1) = *(const u32x4*)(aptr + k0 + 16 + koff);
    if (kc + 1 < NCHUNK) __builtin_prefetch(aptr + k0 + 32, 0, 1);

    // ---- B fragment: one spatial tap, contiguous channels ----
    const int s       = k0 / CIN;
    const int cinBase = k0 - s * CIN;
    const int kh = s / KW;
    const int kw = s - kh * KW;
    const int ih = hRow + kh - PAD;
    const int iw = wCol + kw - PAD;
    v16h bfrag;
    if ((unsigned)ih < (unsigned)HH && (unsigned)iw < (unsigned)WW) {
      const _Float16* bptr =
          in + (size_t)((bImg * HH + ih) * WW + iw) * CIN + cinBase;
      *(u32x4*)&bfrag         = *(const u32x4*)(bptr + koff);
      *(((u32x4*)&bfrag) + 1) = *(const u32x4*)(bptr + 16 + koff);
    } else {
#pragma unroll
      for (int i = 0; i < 16; ++i) bfrag[i] = (_Float16)0.0f;
    }

    acc = __builtin_amdgcn_wmma_f32_16x16x32_f16(false, a, false, bfrag,
                                                 (short)0, acc, false, false);
  }

  // ---- epilogue: 8 consecutive channels per lane, contiguous NHWC store ----
  const int mbase = (lane >> 4) << 3;
  const size_t obase =
      (size_t)((bImg * HH + hRow) * WW + wCol) * CoutTotal + coBase + coTile + mbase;
  if constexpr (sizeof(Tout) == 2) {
    __align__(16) _Float16 hv[8];
#pragma unroll
    for (int r = 0; r < 8; ++r) {
      float v = acc[r] + bias[coTile + mbase + r];
      if (RELU) v = fmaxf(v, 0.0f);
      hv[r] = (_Float16)v;
    }
    *(u32x4*)(out + obase) = *(const u32x4*)hv;
  } else {
    __align__(32) float fv[8];
#pragma unroll
    for (int r = 0; r < 8; ++r) {
      float v = acc[r] + bias[coTile + mbase + r];
      if (RELU) v = fmaxf(v, 0.0f);
      fv[r] = v;
    }
    *(u32x4*)(out + obase)       = ((const u32x4*)fv)[0];
    *(((u32x4*)(out + obase)) + 1) = ((const u32x4*)fv)[1];
  }
}

// ======== small conv (conv1, CIN=2): LDS-staged im2col, NHWC in/out ========
template <int CIN, int KH, int KW, int PAD, bool RELU>
__global__ void conv_small_kernel(const _Float16* __restrict__ in,
                                  const _Float16* __restrict__ wgt,
                                  const float* __restrict__ bias,
                                  _Float16* __restrict__ out,
                                  int Cout, int CoutTotal, int coBase) {
  constexpr int KDIM = CIN * KH * KW;
  constexpr int NCHUNK = (KDIM + 31) / 32;

  __shared__ __align__(16) _Float16 Al[16 * 32];
  __shared__ __align__(16) _Float16 Bt[64 * 32];

  const int wv   = threadIdx.x >> 5;
  const int lane = threadIdx.x & 31;
  const int p0   = blockIdx.x * 64;
  const int bImg = p0 >> 16;
  const int hRow = (p0 >> 8) & 255;
  const int w0   = p0 & 255;
  const int coTile = blockIdx.y * 16;

  v8f acc;
#pragma unroll
  for (int i = 0; i < 8; ++i) acc[i] = 0.0f;

  for (int kc = 0; kc < NCHUNK; ++kc) {
    const int k0 = kc * 32;
    for (int i = threadIdx.x; i < 16 * 32; i += 128) {
      const int mm = i >> 5;
      const int k = (i & 31) + k0;
      _Float16 v = (_Float16)0.0f;
      if (k < KDIM && (coTile + mm) < Cout) v = wgt[(size_t)(coTile + mm) * KDIM + k];
      Al[i] = v;
    }
    {
      const int n   = threadIdx.x >> 1;
      const int kh0 = (threadIdx.x & 1) * 16;
      const int wPix = w0 + n;
#pragma unroll 4
      for (int j = 0; j < 16; ++j) {
        const int k = k0 + kh0 + j;
        _Float16 v = (_Float16)0.0f;
        if (k < KDIM) {
          const int s   = k / CIN;
          const int cin = k - s * CIN;
          const int kh  = s / KW;
          const int kw  = s - kh * KW;
          const int ih  = hRow + kh - PAD;
          const int iw  = wPix + kw - PAD;
          if ((unsigned)ih < (unsigned)HH && (unsigned)iw < (unsigned)WW)
            v = in[(size_t)((bImg * HH + ih) * WW + iw) * CIN + cin];
        }
        Bt[n * 32 + kh0 + j] = v;
      }
    }
    __syncthreads();
    const int mm   = lane & 15;
    const int koff = (lane >> 4) << 3;
    v16h a, bfrag;
    *(u32x4*)&a             = *(const u32x4*)&Al[mm * 32 + koff];
    *(((u32x4*)&a) + 1)     = *(const u32x4*)&Al[mm * 32 + 16 + koff];
    const int n = (wv << 4) | mm;
    *(u32x4*)&bfrag         = *(const u32x4*)&Bt[n * 32 + koff];
    *(((u32x4*)&bfrag) + 1) = *(const u32x4*)&Bt[n * 32 + 16 + koff];
    acc = __builtin_amdgcn_wmma_f32_16x16x32_f16(false, a, false, bfrag,
                                                 (short)0, acc, false, false);
    __syncthreads();
  }

  const int nn    = (wv << 4) | (lane & 15);
  const int wPix  = w0 + nn;
  const int mbase = (lane >> 4) << 3;
#pragma unroll
  for (int r = 0; r < 8; ++r) {
    const int co = coTile + mbase + r;
    if (co < Cout) {
      float v = acc[r] + bias[co];
      if (RELU) v = fmaxf(v, 0.0f);
      out[(size_t)((bImg * HH + hRow) * WW + wPix) * CoutTotal + coBase + co] =
          (_Float16)v;
    }
  }
}

// ================== generic linear (x @ W^T + b) via WMMA ==================
__global__ void linear_wmma_kernel(const float* __restrict__ in,
                                   const float* __restrict__ w,
                                   const float* __restrict__ bias,
                                   float* __restrict__ out,
                                   int T, int Din, int Dout, int relu) {
  __shared__ __align__(16) _Float16 Al[16 * 32];
  __shared__ __align__(16) _Float16 Bt[64 * 32];

  const int wv   = threadIdx.x >> 5;
  const int lane = threadIdx.x & 31;
  const int t0     = blockIdx.x * 64;
  const int coTile = blockIdx.y * 16;
  const int nchunk = (Din + 31) / 32;

  v8f acc;
#pragma unroll
  for (int i = 0; i < 8; ++i) acc[i] = 0.0f;

  for (int kc = 0; kc < nchunk; ++kc) {
    const int k0 = kc * 32;
    for (int i = threadIdx.x; i < 16 * 32; i += 128) {
      const int m = i >> 5;
      const int k = (i & 31) + k0;
      const int co = coTile + m;
      _Float16 v = (_Float16)0.0f;
      if (k < Din && co < Dout) v = (_Float16)w[(size_t)co * Din + k];
      Al[i] = v;
    }
    {
      const int n   = threadIdx.x >> 1;
      const int kh0 = (threadIdx.x & 1) * 16;
#pragma unroll 4
      for (int j = 0; j < 16; ++j) {
        const int k = k0 + kh0 + j;
        _Float16 v = (_Float16)0.0f;
        if (k < Din) v = (_Float16)in[(size_t)(t0 + n) * Din + k];
        Bt[n * 32 + kh0 + j] = v;
      }
    }
    __syncthreads();

    const int m    = lane & 15;
    const int koff = (lane >> 4) << 3;
    v16h a, bfrag;
    *(u32x4*)&a             = *(const u32x4*)&Al[m * 32 + koff];
    *(((u32x4*)&a) + 1)     = *(const u32x4*)&Al[m * 32 + 16 + koff];
    const int n = (wv << 4) | m;
    *(u32x4*)&bfrag         = *(const u32x4*)&Bt[n * 32 + koff];
    *(((u32x4*)&bfrag) + 1) = *(const u32x4*)&Bt[n * 32 + 16 + koff];

    acc = __builtin_amdgcn_wmma_f32_16x16x32_f16(false, a, false, bfrag,
                                                 (short)0, acc, false, false);
    __syncthreads();
  }

  const int tok   = t0 + ((wv << 4) | (lane & 15));
  const int mbase = (lane >> 4) << 3;
#pragma unroll
  for (int r = 0; r < 8; ++r) {
    const int co = coTile + mbase + r;
    if (co < Dout) {
      float v = acc[r] + bias[co];
      if (relu) v = fmaxf(v, 0.0f);
      out[(size_t)tok * Dout + co] = v;
    }
  }
}

// ============== integral image over NHWC feat (fp32, 2 passes) =============
__global__ void integral_rows_kernel(const float* __restrict__ feat,
                                     float* __restrict__ S) {
  const int bc = blockIdx.x;                       // b*128 + c
  const int bImg = bc >> 7;
  const int c = bc & 127;
  float* sp = S + (size_t)bc * SPLANE;
  for (int w = threadIdx.x; w <= WW; w += 256) sp[w] = 0.0f;  // top row
  const int h = threadIdx.x;                       // 256 threads = rows
  sp[(h + 1) * SW] = 0.0f;                         // left column
  float acc = 0.0f;
  for (int w = 0; w < WW; ++w) {
    acc += feat[(size_t)((bImg * HH + h) * WW + w) * HID + c];
    sp[(h + 1) * SW + (w + 1)] = acc;
  }
}

__global__ void integral_cols_kernel(float* __restrict__ S) {
  const int bc = blockIdx.x;
  float* sp = S + (size_t)bc * SPLANE;
  const int col = threadIdx.x + 1;                 // columns 1..256
  float acc = 0.0f;
  for (int h = 1; h <= HH; ++h) {
    acc += sp[h * SW + col];
    sp[h * SW + col] = acc;
  }
}

// ============================== ROI mean pool ==============================
__global__ void roipool_kernel(const float* __restrict__ S,
                               const float* __restrict__ bboxes,
                               float* __restrict__ pooled) {
  const int bn = blockIdx.x;                       // b*NOBJ + n
  const int b = bn >> 7;
  const float cx = bboxes[bn * 4 + 0], cy = bboxes[bn * 4 + 1];
  const float bw = bboxes[bn * 4 + 2], bh = bboxes[bn * 4 + 3];
  int x1 = (int)((cx - bw * 0.5f) * (float)WW);    // trunc toward zero
  int y1 = (int)((cy - bh * 0.5f) * (float)HH);
  int x2 = (int)((cx + bw * 0.5f) * (float)WW);
  int y2 = (int)((cy + bh * 0.5f) * (float)HH);
  x1 = min(max(x1, 0), WW); x2 = min(max(x2, 0), WW);
  y1 = min(max(y1, 0), HH); y2 = min(max(y2, 0), HH);
  const bool nonempty = (x2 > x1) && (y2 > y1);
  const float inva = 1.0f / fmaxf((float)((x2 - x1) * (y2 - y1)), 1.0f);
  const int c = threadIdx.x;                       // 128 channels
  const float* sp = S + ((size_t)(b * HID + c)) * SPLANE;
  float v = 0.0f;
  if (nonempty)
    v = (sp[y2 * SW + x2] - sp[y1 * SW + x2] - sp[y2 * SW + x1] + sp[y1 * SW + x1]) * inva;
  pooled[(size_t)bn * HID + c] = v;
}

// ============================ small elementwise ============================
__global__ void concat_kernel(const float* __restrict__ a,
                              const float* __restrict__ b,
                              float* __restrict__ out) {  // [T][256]
  const int t = blockIdx.x, d = threadIdx.x;              // 256 threads
  out[(size_t)t * 256 + d] =
      (d < HID) ? a[(size_t)t * HID + d] : b[(size_t)t * HID + (d - HID)];
}

__global__ void add_bcast_kernel(float* __restrict__ x,
                                 const float* __restrict__ pe) {
  x[(size_t)blockIdx.x * HID + threadIdx.x] += pe[threadIdx.x];
}

__global__ void sinpe_kernel(float* __restrict__ x) {
  const int t = blockIdx.x, d = threadIdx.x;
  const int n = t & (NOBJ - 1);
  const int j = d >> 1;
  const float div = __expf((float)(2 * j) * (-9.210340371976184f / (float)HID));
  const float ang = (float)n * div;
  x[(size_t)t * HID + d] += (d & 1) ? __cosf(ang) : __sinf(ang);
}

// ===================== attention: per-(b,h) online softmax =================
__global__ void attention_kernel(const float* __restrict__ qkv,  // [T][384]
                                 float* __restrict__ o) {        // [T][128]
  __shared__ float Kl[NOBJ * 16];
  __shared__ float Vl[NOBJ * 16];
  const int b = blockIdx.x >> 3;
  const int h = blockIdx.x & 7;
  const int q = threadIdx.x;                                     // 128 threads
  const float* base = qkv + (size_t)(b * NOBJ + q) * (3 * HID);
  float qv[16];
#pragma unroll
  for (int d = 0; d < 16; ++d) {
    qv[d] = base[h * 16 + d];
    Kl[q * 16 + d] = base[HID + h * 16 + d];
    Vl[q * 16 + d] = base[2 * HID + h * 16 + d];
  }
  __syncthreads();
  float mx = -1e30f, l = 0.0f, oacc[16];
#pragma unroll
  for (int d = 0; d < 16; ++d) oacc[d] = 0.0f;
  for (int kk = 0; kk < NOBJ; ++kk) {
    float s = 0.0f;
#pragma unroll
    for (int d = 0; d < 16; ++d) s += qv[d] * Kl[kk * 16 + d];
    s *= 0.25f;                                                  // 1/sqrt(16)
    const float nm = fmaxf(mx, s);
    const float corr = __expf(mx - nm);
    const float pp = __expf(s - nm);
    l = l * corr + pp;
#pragma unroll
    for (int d = 0; d < 16; ++d) oacc[d] = oacc[d] * corr + pp * Vl[kk * 16 + d];
    mx = nm;
  }
  const float inv = 1.0f / l;
#pragma unroll
  for (int d = 0; d < 16; ++d)
    o[(size_t)(b * NOBJ + q) * HID + h * 16 + d] = oacc[d] * inv;
}

// ======================= residual add + LayerNorm (D=128) ==================
__global__ void add_ln_kernel(float* __restrict__ x, const float* __restrict__ y,
                              const float* __restrict__ g,
                              const float* __restrict__ bta) {
  __shared__ float red[HID];
  const int t = blockIdx.x, d = threadIdx.x;
  const size_t idx = (size_t)t * HID + d;
  const float v = x[idx] + y[idx];
  red[d] = v; __syncthreads();
  for (int s = 64; s > 0; s >>= 1) { if (d < s) red[d] += red[d + s]; __syncthreads(); }
  const float mean = red[0] * (1.0f / HID);
  __syncthreads();
  const float c = v - mean;
  red[d] = c * c; __syncthreads();
  for (int s = 64; s > 0; s >>= 1) { if (d < s) red[d] += red[d + s]; __syncthreads(); }
  const float var = red[0] * (1.0f / HID);
  x[idx] = c * rsqrtf(var + 1e-5f) * g[d] + bta[d];
}

// =============================== mask / output =============================
__global__ void mask_mul_kernel(const float* __restrict__ t,
                                const unsigned char* __restrict__ mask,
                                float* __restrict__ attf) {
  const int tok = blockIdx.x, d = threadIdx.x;
  const float m = mask[tok] ? 1.0f : 0.0f;
  attf[(size_t)tok * HID + d] = t[(size_t)tok * HID + d] * m;
}

__global__ void assemble_kernel(const float* __restrict__ prev,
                                const unsigned char* __restrict__ mask,
                                const float* __restrict__ attf,
                                const float* __restrict__ olog,
                                const float* __restrict__ clog,
                                const float* __restrict__ tlog,
                                float* __restrict__ out) {
  const int t = blockIdx.x, d = threadIdx.x;
  out[5120 + (size_t)t * HID + d] = attf[(size_t)t * HID + d];
  const float m = mask[t] ? 1.0f : 0.0f;
  if (d < 4) {
    const float off = olog[t * 4 + d];
    out[t * 4 + d]        = fminf(fmaxf(prev[t * 4 + d] + off, 0.0f), 1.0f) * m;
    out[2048 + t * 4 + d] = off * m;
  }
  if (d == 0) {
    out[4096 + t] = (1.0f / (1.0f + __expf(-clog[t]))) * m;
    out[4608 + t] = (1.0f / (1.0f + __expf(-tlog[t]))) * m;
  }
}

// ================================= driver ==================================
static inline void run_linear(const float* in, const float* w, const float* b,
                              float* out, int T, int Din, int Dout, int relu,
                              hipStream_t s) {
  dim3 g(T / 64, (Dout + 15) / 16);
  linear_wmma_kernel<<<g, 128, 0, s>>>(in, w, b, out, T, Din, Dout, relu);
}

extern "C" void kernel_launch(void* const* d_in, const int* in_sizes, int n_in,
                              void* d_out, int out_size, void* d_ws, size_t ws_size,
                              hipStream_t stream) {
  (void)in_sizes; (void)out_size; (void)ws_size;

  // ---- input map (setup_inputs dict order) ----
  const float* mv   = (const float*)d_in[0];           // [4,2,256,256]
  const float* bb   = (const float*)d_in[1];           // [4,128,4]
  const unsigned char* mask = (const unsigned char*)d_in[2];  // bool [4,128]
  const float* P[64];
  for (int i = 3; i < n_in && i < 64 + 3; ++i) P[i - 3] = (const float*)d_in[i];
  // 0 conv1_w 1 conv1_b 2 conv2_w 3 conv2_b 4 conv3_w 5 conv3_b
  // 6 s1_w 7 s1_b 8 s2_w 9 s2_b 10 s3_w 11 s3_b 12 fus_w 13 fus_b
  // 14 be1_w 15 be1_b 16 be2_w 17 be2_b 18 pos_emb
  // 19+l*12: qkv_w qkv_b out_w out_b ln1_g ln1_b ff1_w ff1_b ff2_w ff2_b ln2_g ln2_b
  // 43 fc1_w 44 fc1_b 45 fc2_w 46 fc2_b 47 off1_w 48 off1_b 49 off2_w 50 off2_b
  // 51 cf1_w 52 cf1_b 53 cf2_w 54 cf2_b 55 tm1_w 56 tm1_b 57 tm2_w 58 tm2_b

  // ---- workspace carve ----
  size_t off = 0;
  auto carve = [&](size_t bytes) -> void* {
    off = (off + 255) & ~(size_t)255;
    void* p = (char*)d_ws + off;
    off += bytes;
    return p;
  };
  const size_t NPIX = (size_t)BATCH * HH * WW;         // 262144
  _Float16* wc1 = (_Float16*)carve(sizeof(_Float16) * 32 * 18);
  _Float16* wc2 = (_Float16*)carve(sizeof(_Float16) * 64 * 288);
  _Float16* wc3 = (_Float16*)carve(sizeof(_Float16) * 128 * 576);
  _Float16* ws1 = (_Float16*)carve(sizeof(_Float16) * 64 * 128);
  _Float16* ws2 = (_Float16*)carve(sizeof(_Float16) * 64 * 1152);
  _Float16* ws3 = (_Float16*)carve(sizeof(_Float16) * 64 * 3200);
  _Float16* wfu = (_Float16*)carve(sizeof(_Float16) * 128 * 192);
  _Float16* mvh = (_Float16*)carve(sizeof(_Float16) * NPIX * 2);
  _Float16* x1  = (_Float16*)carve(sizeof(_Float16) * NPIX * 32);
  _Float16* x2  = (_Float16*)carve(sizeof(_Float16) * NPIX * 64);
  _Float16* x3  = (_Float16*)carve(sizeof(_Float16) * NPIX * 128);
  _Float16* ms  = (_Float16*)carve(sizeof(_Float16) * NPIX * 192);
  float* feat   = (float*)carve(sizeof(float) * NPIX * HID);
  float* Simg   = (float*)carve(sizeof(float) * (size_t)BATCH * HID * SPLANE);
  float* h64    = (float*)carve(sizeof(float) * NTOK * 64);
  float* objb   = (float*)carve(sizeof(float) * NTOK * HID);
  float* pooled = (float*)carve(sizeof(float) * NTOK * HID);
  float* comb   = (float*)carve(sizeof(float) * NTOK * 256);
  float* f1     = (float*)carve(sizeof(float) * NTOK * HID);
  float* tbuf   = (float*)carve(sizeof(float) * NTOK * HID);
  float* qkvb   = (float*)carve(sizeof(float) * NTOK * 3 * HID);
  float* attnb  = (float*)carve(sizeof(float) * NTOK * HID);
  float* projb  = (float*)carve(sizeof(float) * NTOK * HID);
  float* ffb    = (float*)carve(sizeof(float) * NTOK * FFD);
  float* ff2b   = (float*)carve(sizeof(float) * NTOK * HID);
  float* attf   = (float*)carve(sizeof(float) * NTOK * HID);
  float* oh     = (float*)carve(sizeof(float) * NTOK * 64);
  float* olog   = (float*)carve(sizeof(float) * NTOK * 4);
  float* ch     = (float*)carve(sizeof(float) * NTOK * 32);
  float* clog   = (float*)carve(sizeof(float) * NTOK);
  float* th     = (float*)carve(sizeof(float) * NTOK * 32);
  float* tlog   = (float*)carve(sizeof(float) * NTOK);

  // ---- weight reorder/convert to f16, k = (kh*KW+kw)*CIN + cin ----
  auto cvtw = [&](const float* src, _Float16* dst, int Cout, int CIN, int KHW) {
    const int n = Cout * CIN * KHW;
    cvt_w_kernel<<<(n + 255) / 256, 256, 0, stream>>>(src, dst, Cout, CIN, KHW);
  };
  cvtw(P[0],  wc1, 32, 2, 9);
  cvtw(P[2],  wc2, 64, 32, 9);
  cvtw(P[4],  wc3, 128, 64, 9);
  cvtw(P[6],  ws1, 64, 128, 1);
  cvtw(P[8],  ws2, 64, 128, 9);
  cvtw(P[10], ws3, 64, 128, 25);
  cvtw(P[12], wfu, 128, 192, 1);

  // ---- motion encoder ----
  mv_to_nhwc_kernel<<<(int)(NPIX / 256), 256, 0, stream>>>(mv, mvh);
  const int NPIXT = (int)(NPIX / 64);                  // 4096
  conv_small_kernel<2, 3, 3, 1, true>
      <<<dim3(NPIXT, 32 / 16), 128, 0, stream>>>(mvh, wc1, P[1], x1, 32, 32, 0);
  conv_fast_kernel<32, 3, 3, 1, true, _Float16>
      <<<dim3(NPIXT, 64 / 16), 128, 0, stream>>>(x1, wc2, P[3], x2, 64, 0);
  conv_fast_kernel<64, 3, 3, 1, true, _Float16>
      <<<dim3(NPIXT, 128 / 16), 128, 0, stream>>>(x2, wc3, P[5], x3, 128, 0);
  conv_fast_kernel<128, 1, 1, 0, true, _Float16>
      <<<dim3(NPIXT, 64 / 16), 128, 0, stream>>>(x3, ws1, P[7], ms, 192, 0);
  conv_fast_kernel<128, 3, 3, 1, true, _Float16>
      <<<dim3(NPIXT, 64 / 16), 128, 0, stream>>>(x3, ws2, P[9], ms, 192, 64);
  conv_fast_kernel<128, 5, 5, 2, true, _Float16>
      <<<dim3(NPIXT, 64 / 16), 128, 0, stream>>>(x3, ws3, P[11], ms, 192, 128);
  conv_fast_kernel<192, 1, 1, 0, false, float>
      <<<dim3(NPIXT, 128 / 16), 128, 0, stream>>>(ms, wfu, P[13], feat, 128, 0);

  // ---- integral image + ROI pooling ----
  integral_rows_kernel<<<BATCH * HID, 256, 0, stream>>>(feat, Simg);
  integral_cols_kernel<<<BATCH * HID, 256, 0, stream>>>(Simg);
  roipool_kernel<<<NTOK, HID, 0, stream>>>(Simg, bb, pooled);

  // ---- object encoder ----
  run_linear(bb,  P[14], P[15], h64,  NTOK, 4,  64,  1, stream);
  run_linear(h64, P[16], P[17], objb, NTOK, 64, HID, 1, stream);
  add_bcast_kernel<<<NTOK, HID, 0, stream>>>(objb, P[18]);  // + pos_emb

  // ---- combine + MLP ----
  concat_kernel<<<NTOK, 256, 0, stream>>>(objb, pooled, comb);
  run_linear(comb, P[43], P[44], f1,   NTOK, 2 * HID, HID, 1, stream);
  run_linear(f1,   P[45], P[46], tbuf, NTOK, HID,     HID, 1, stream);
  sinpe_kernel<<<NTOK, HID, 0, stream>>>(tbuf);

  // ---- transformer (post-LN, 2 layers) ----
  for (int l = 0; l < 2; ++l) {
    const int pb = 19 + l * 12;
    run_linear(tbuf, P[pb + 0], P[pb + 1], qkvb, NTOK, HID, 3 * HID, 0, stream);
    attention_kernel<<<BATCH * HEADS, NOBJ, 0, stream>>>(qkvb, attnb);
    run_linear(attnb, P[pb + 2], P[pb + 3], projb, NTOK, HID, HID, 0, stream);
    add_ln_kernel<<<NTOK, HID, 0, stream>>>(tbuf, projb, P[pb + 4], P[pb + 5]);
    run_linear(tbuf, P[pb + 6], P[pb + 7], ffb,  NTOK, HID, FFD, 1, stream);
    run_linear(ffb,  P[pb + 8], P[pb + 9], ff2b, NTOK, FFD, HID, 0, stream);
    add_ln_kernel<<<NTOK, HID, 0, stream>>>(tbuf, ff2b, P[pb + 10], P[pb + 11]);
  }

  // ---- heads ----
  mask_mul_kernel<<<NTOK, HID, 0, stream>>>(tbuf, mask, attf);
  run_linear(attf, P[47], P[48], oh,   NTOK, HID, 64, 1, stream);
  run_linear(oh,   P[49], P[50], olog, NTOK, 64,  4,  0, stream);
  run_linear(attf, P[51], P[52], ch,   NTOK, HID, 32, 1, stream);
  run_linear(ch,   P[53], P[54], clog, NTOK, 32,  1,  0, stream);
  run_linear(attf, P[55], P[56], th,   NTOK, HID, 32, 1, stream);
  run_linear(th,   P[57], P[58], tlog, NTOK, 32,  1,  0, stream);

  assemble_kernel<<<NTOK, HID, 0, stream>>>(bb, mask, attf, olog, clog, tlog,
                                            (float*)d_out);
}
